// Optimized_MHSA_21328807592208
// MI455X (gfx1250) — matrix operand order repaired and checked by reading, not compiled, not run on hardware
//
#include <hip/hip_runtime.h>
#include <hip/hip_bf16.h>

#define DM   1024
#define NH   16
#define HD   64
#define SEQ  2048
#define BATCH 2

typedef __attribute__((ext_vector_type(16))) __bf16 v16bf;
typedef __attribute__((ext_vector_type(8)))  float  v8f;

__device__ __forceinline__ __bf16 f2bf(float f) { return (__bf16)f; }

__device__ __forceinline__ v8f wmma_bf16(v16bf a, v16bf b, v8f c) {
  // D = A(16x32 bf16) * B(32x16 bf16) + C(16x16 f32)
  return __builtin_amdgcn_wmma_f32_16x16x32_bf16(false, a, false, b, (short)0, c,
                                                 false, false);
}

// CDNA5 async copy: 16B global -> LDS, tracked on ASYNCcnt (CDNA5 ISA).
__device__ __forceinline__ void async_copy_b128(unsigned lds_dst, const void* gsrc) {
  asm volatile("global_load_async_to_lds_b128 %0, %1, off"
               :: "v"(lds_dst), "v"((unsigned long long)(uintptr_t)gsrc)
               : "memory");
}
__device__ __forceinline__ void wait_asynccnt0() {
  asm volatile("s_wait_asynccnt 0" ::: "memory");
}
// Generic LDS pointer: low 32 bits are the LDS byte offset.
__device__ __forceinline__ unsigned lds_off(const void* p) {
  return (unsigned)(uintptr_t)p;
}

// ---------------- fragment loaders (contiguous k = 16*half + i, same in A and B) --

// A 16x32 from row-major bf16, ld in elements.
__device__ __forceinline__ v16bf load_a_bf16(const __bf16* src, int ld, int lane) {
  const int half = lane >> 4, lr = lane & 15;
  const __bf16* p0 = src + lr * ld + half * 16;
  v16bf a;
#pragma unroll
  for (int i = 0; i < 16; ++i) a[i] = p0[i];
  return a;
}

// B 32x16 where the source is stored [N][K] row-major (B[k][n] = src[n*ld+k]):
// contiguous 32B per lane.
__device__ __forceinline__ v16bf load_b_nk_bf16(const __bf16* src, int ld, int lane) {
  const int half = lane >> 4, lr = lane & 15;
  const __bf16* p0 = src + lr * ld + half * 16;
  v16bf b;
#pragma unroll
  for (int i = 0; i < 16; ++i) b[i] = p0[i];
  return b;
}

// B 32x16 where the source is stored [K][N] row-major (strided per lane).
__device__ __forceinline__ v16bf load_b_kn_bf16(const __bf16* src, int ld, int lane) {
  const int half = lane >> 4, lr = lane & 15;
  const __bf16* p0 = src + half * 16 * ld + lr;
  v16bf b;
#pragma unroll
  for (int i = 0; i < 16; ++i) b[i] = p0[i * ld];
  return b;
}

// ---------------- kernel 0: one-shot f32 -> bf16 conversion --------------------
__global__ void cvt_f32_bf16_kernel(const float* __restrict__ src,
                                    __bf16* __restrict__ dst, int n) {
  const int i = (blockIdx.x * blockDim.x + threadIdx.x) * 4;
  if (i < n) {
    float4 f = *(const float4*)(src + i);
    dst[i + 0] = f2bf(f.x);
    dst[i + 1] = f2bf(f.y);
    dst[i + 2] = f2bf(f.z);
    dst[i + 3] = f2bf(f.w);
  }
}

// ---------------- kernel 1: h = x @ W_in^T + b_in ------------------------------
// Emits h in [b,head,s,64] AND transposed hT in [b,head,64,s] (for the V operand).
// grid: (DM/64, 4096/128); block 256 (8 waves).
__global__ void gemm_in_kernel(const __bf16* __restrict__ xbf,
                               const __bf16* __restrict__ Wbf,
                               const float* __restrict__ bias,
                               __bf16* __restrict__ hout,
                               __bf16* __restrict__ houtT) {
  __shared__ __bf16 bt[2][64 * 32];  // double-buffered W tile [n 64][k 32]
  const int tid = threadIdx.x;
  const int wave = tid >> 5, lane = tid & 31;
  const int half = lane >> 4, lr = lane & 15;
  const int m0 = blockIdx.y * 128 + wave * 16;
  const int n0 = blockIdx.x * 64;

  const int srow = tid >> 2;          // 0..63
  const int scol = (tid & 3) * 8;     // 0,8,16,24
  const __bf16* wsrc = Wbf + (size_t)(n0 + srow) * DM + scol;
  const __bf16* arow = xbf + (size_t)(m0 + lr) * DM;

  async_copy_b128(lds_off(&bt[0][srow * 32 + scol]), wsrc);  // prologue: k0 = 0

  v8f c[4] = {};
  for (int k0 = 0, it = 0; k0 < DM; k0 += 32, ++it) {
    const int cur = it & 1;
    wait_asynccnt0();
    __syncthreads();
    if (k0 + 32 < DM) {
      async_copy_b128(lds_off(&bt[1 - cur][srow * 32 + scol]), wsrc + k0 + 32);
      __builtin_prefetch(arow + k0 + 32, 0, 1);
    }
    v16bf a = load_a_bf16(xbf + (size_t)m0 * DM + k0, DM, lane);
#pragma unroll
    for (int j = 0; j < 4; ++j) {
      v16bf b = load_b_nk_bf16(&bt[cur][(j * 16) * 32], 32, lane);
      c[j] = wmma_bf16(a, b, c[j]);
    }
  }
#pragma unroll
  for (int j = 0; j < 4; ++j)
#pragma unroll
    for (int r = 0; r < 8; ++r) {
      const int m = m0 + half * 8 + r;
      const int n = n0 + j * 16 + lr;
      const __bf16 v = f2bf(c[j][r] + bias[n]);
      const int bb = m >> 11, s = m & (SEQ - 1);
      const int hh = n >> 6, d = n & (HD - 1);
      const size_t bhh = (size_t)(bb * NH + hh);
      hout[(bhh * SEQ + s) * HD + d] = v;
      houtT[(bhh * HD + d) * SEQ + s] = v;
    }
}

// ---------------- kernel 2: q = (h @ attn_w[head]) * (log2e/64), bf16 ----------
__global__ void qproj_kernel(const __bf16* __restrict__ hbuf,
                             const __bf16* __restrict__ awbf,
                             __bf16* __restrict__ qbuf) {
  const int tid = threadIdx.x;
  const int wave = tid >> 5, lane = tid & 31;
  const int half = lane >> 4, lr = lane & 15;
  const int bh = blockIdx.y;
  const int head = bh & (NH - 1);
  const int m0 = blockIdx.x * 128 + wave * 16;

  const __bf16* hB = hbuf + (size_t)bh * SEQ * HD;
  __bf16* qB = qbuf + (size_t)bh * SEQ * HD;
  const __bf16* w = awbf + (size_t)head * HD * HD;  // [d][e] row-major

  v8f c[4] = {};
#pragma unroll
  for (int k0 = 0; k0 < HD; k0 += 32) {
    v16bf a = load_a_bf16(hB + (size_t)m0 * HD + k0, HD, lane);
#pragma unroll
    for (int j = 0; j < 4; ++j) {
      v16bf b = load_b_kn_bf16(w + (size_t)k0 * HD + j * 16, HD, lane);
      c[j] = wmma_bf16(a, b, c[j]);
    }
  }
  // fold 1/HD score scale AND log2(e) so flash softmax can use bare v_exp_f32
  const float scale = 0x1.715476p+0f / (float)HD;
#pragma unroll
  for (int j = 0; j < 4; ++j)
#pragma unroll
    for (int r = 0; r < 8; ++r) {
      const int m = m0 + half * 8 + r;
      const int n = j * 16 + lr;
      qB[(size_t)m * HD + n] = f2bf(c[j][r] * scale);
    }
}

// ---------------- kernel 3: flash attention per (b,head) -----------------------
// grid: (SEQ/128, BATCH*NH); block 256 (8 waves); wave = 16-query strip.
__global__ void flash_attn_kernel(const __bf16* __restrict__ hbuf,
                                  const __bf16* __restrict__ hTbuf,
                                  const __bf16* __restrict__ qbuf,
                                  __bf16* __restrict__ obuf) {
  __shared__ __bf16 kv[2][32 * 64];   // K tile  [key][d]  (2 x 4KB)
  __shared__ __bf16 kvT[2][64 * 32];  // V tile  [d][key]  (2 x 4KB)
  __shared__ __bf16 pbuf[8][16 * 32]; // per-wave P repack scratch (8KB)

  const int tid = threadIdx.x;
  const int wave = tid >> 5, lane = tid & 31;
  const int half = lane >> 4, lr = lane & 15;
  const int bh = blockIdx.y;
  const int b = bh >> 4, head = bh & (NH - 1);
  const int m0 = blockIdx.x * 128 + wave * 16;

  const __bf16* hB = hbuf + (size_t)bh * SEQ * HD;
  const __bf16* hTB = hTbuf + (size_t)bh * HD * SEQ;
  const __bf16* qB = qbuf + (size_t)bh * SEQ * HD;

  const v16bf qa0 = load_a_bf16(qB + (size_t)m0 * HD, HD, lane);
  const v16bf qa1 = load_a_bf16(qB + (size_t)m0 * HD + 32, HD, lane);

  // staging roles: one 16B async transfer per thread per tile
  const int r1 = tid >> 3, c1 = (tid & 7) * 8;  // kv : row 0..31, col 0..56
  const int r2 = tid >> 2, c2 = (tid & 3) * 8;  // kvT: row 0..63, col 0..24
  const __bf16* src1 = hB + (size_t)r1 * HD + c1;       // + kt*32*HD
  const __bf16* src2 = hTB + (size_t)r2 * SEQ + c2;     // + kt*32

  // prologue: stage tile 0 into buffer 0
  async_copy_b128(lds_off(&kv[0][r1 * 64 + c1]), src1);
  async_copy_b128(lds_off(&kvT[0][r2 * 32 + c2]), src2);

  v8f o[4] = {};
  float row_max[8], row_sum[8];
#pragma unroll
  for (int r = 0; r < 8; ++r) { row_max[r] = -INFINITY; row_sum[r] = 0.0f; }

  for (int kt = 0; kt < SEQ / 32; ++kt) {
    const int cur = kt & 1;
    wait_asynccnt0();      // my transfers for tile kt done
    __syncthreads();       // everyone's done; other buffer free to overwrite
    if (kt + 1 < SEQ / 32) {
      async_copy_b128(lds_off(&kv[1 - cur][r1 * 64 + c1]),
                      src1 + (size_t)(kt + 1) * 32 * HD);
      async_copy_b128(lds_off(&kvT[1 - cur][r2 * 32 + c2]),
                      src2 + (size_t)(kt + 1) * 32);
    }

    // scores: S = Q @ K^T (log2e/64 scale already folded into Q)
    v8f s[2];
#pragma unroll
    for (int j = 0; j < 2; ++j) {
      v8f acc = {};
      acc = wmma_bf16(qa0, load_b_nk_bf16(&kv[cur][(j * 16) * 64], 64, lane), acc);
      acc = wmma_bf16(qa1, load_b_nk_bf16(&kv[cur][(j * 16) * 64 + 32], 64, lane), acc);
      s[j] = acc;
    }

    // online softmax in exp2 domain; rows at (half*8+r), cols span a 16-lane half
    float nm[8], alpha[8];
#pragma unroll
    for (int r = 0; r < 8; ++r) {
      float v = fmaxf(s[0][r], s[1][r]);
      v = fmaxf(v, __shfl_xor(v, 1, 32));
      v = fmaxf(v, __shfl_xor(v, 2, 32));
      v = fmaxf(v, __shfl_xor(v, 4, 32));
      v = fmaxf(v, __shfl_xor(v, 8, 32));
      nm[r] = fmaxf(row_max[r], v);
      alpha[r] = exp2f(row_max[r] - nm[r]);
      row_max[r] = nm[r];
    }
#pragma unroll
    for (int j = 0; j < 2; ++j)
#pragma unroll
      for (int r = 0; r < 8; ++r) s[j][r] = exp2f(s[j][r] - nm[r]);
#pragma unroll
    for (int r = 0; r < 8; ++r) {
      float sum = s[0][r] + s[1][r];
      sum += __shfl_xor(sum, 1, 32);
      sum += __shfl_xor(sum, 2, 32);
      sum += __shfl_xor(sum, 4, 32);
      sum += __shfl_xor(sum, 8, 32);
      row_sum[r] = row_sum[r] * alpha[r] + sum;
    }
#pragma unroll
    for (int j = 0; j < 4; ++j)
#pragma unroll
      for (int r = 0; r < 8; ++r) o[j][r] *= alpha[r];

    // repack P (C layout) -> A layout via wave-private LDS
    __bf16* pw = &pbuf[wave][0];
#pragma unroll
    for (int j = 0; j < 2; ++j)
#pragma unroll
      for (int r = 0; r < 8; ++r)
        pw[(half * 8 + r) * 32 + j * 16 + lr] = f2bf(s[j][r]);
    v16bf pa = load_a_bf16(pw, 32, lane);

    // O += P @ V using the transposed V tile: contiguous [N][K] fragment loads
#pragma unroll
    for (int j = 0; j < 4; ++j) {
      v16bf vb = load_b_nk_bf16(&kvT[cur][(j * 16) * 32], 32, lane);
      o[j] = wmma_bf16(pa, vb, o[j]);
    }
  }

  // normalize + store to [b, s, 1024] bf16
#pragma unroll
  for (int j = 0; j < 4; ++j)
#pragma unroll
    for (int r = 0; r < 8; ++r) {
      const int sr = m0 + half * 8 + r;
      const int n = head * HD + j * 16 + lr;
      obuf[((size_t)b * SEQ + sr) * DM + n] = f2bf(o[j][r] / row_sum[r]);
    }
}

// ---------------- kernel 4: out = o @ W_out^T + b_out, f32 ---------------------
__global__ void gemm_out_kernel(const __bf16* __restrict__ obuf,
                                const __bf16* __restrict__ Wbf,
                                const float* __restrict__ bias,
                                float* __restrict__ out) {
  __shared__ __bf16 bt[2][64 * 32];
  const int tid = threadIdx.x;
  const int wave = tid >> 5, lane = tid & 31;
  const int half = lane >> 4, lr = lane & 15;
  const int m0 = blockIdx.y * 128 + wave * 16;
  const int n0 = blockIdx.x * 64;

  const int srow = tid >> 2;
  const int scol = (tid & 3) * 8;
  const __bf16* wsrc = Wbf + (size_t)(n0 + srow) * DM + scol;
  const __bf16* arow = obuf + (size_t)(m0 + lr) * DM;

  async_copy_b128(lds_off(&bt[0][srow * 32 + scol]), wsrc);

  v8f c[4] = {};
  for (int k0 = 0, it = 0; k0 < DM; k0 += 32, ++it) {
    const int cur = it & 1;
    wait_asynccnt0();
    __syncthreads();
    if (k0 + 32 < DM) {
      async_copy_b128(lds_off(&bt[1 - cur][srow * 32 + scol]), wsrc + k0 + 32);
      __builtin_prefetch(arow + k0 + 32, 0, 1);
    }
    v16bf a = load_a_bf16(obuf + (size_t)m0 * DM + k0, DM, lane);
#pragma unroll
    for (int j = 0; j < 4; ++j) {
      v16bf b = load_b_nk_bf16(&bt[cur][(j * 16) * 32], 32, lane);
      c[j] = wmma_bf16(a, b, c[j]);
    }
  }
#pragma unroll
  for (int j = 0; j < 4; ++j)
#pragma unroll
    for (int r = 0; r < 8; ++r) {
      const int m = m0 + half * 8 + r;
      const int n = n0 + j * 16 + lr;
      out[(size_t)m * DM + n] = c[j][r] + bias[n];
    }
}

// -------------------------------------------------------------------------------
extern "C" void kernel_launch(void* const* d_in, const int* in_sizes, int n_in,
                              void* d_out, int out_size, void* d_ws, size_t ws_size,
                              hipStream_t stream) {
  (void)in_sizes; (void)n_in; (void)out_size; (void)ws_size;
  const float* x    = (const float*)d_in[0];
  const float* Win  = (const float*)d_in[1];
  const float* bin  = (const float*)d_in[2];
  const float* aw   = (const float*)d_in[3];
  const float* Wout = (const float*)d_in[4];
  const float* bout = (const float*)d_in[5];
  float* out = (float*)d_out;

  const size_t nx = (size_t)BATCH * SEQ * DM;   // 4,194,304
  const size_t nw = (size_t)DM * DM;            // 1,048,576
  const size_t na = (size_t)NH * HD * HD;       //    65,536
  const size_t per = (size_t)BATCH * NH * SEQ * HD;

  __bf16* hbuf   = (__bf16*)d_ws;        // h   [b,head,s,64]
  __bf16* hTbuf  = hbuf + per;           // h^T [b,head,64,s]
  __bf16* qbuf   = hTbuf + per;          // q (scaled)
  __bf16* obuf   = qbuf + per;           // attn out [b,s,1024]
  __bf16* xbf    = obuf + per;           // x bf16
  __bf16* winbf  = xbf + nx;             // W_in bf16
  __bf16* woutbf = winbf + nw;           // W_out bf16
  __bf16* awbf   = woutbf + nw;          // attn_weight bf16

  dim3 blk(256);
  cvt_f32_bf16_kernel<<<dim3((nx / 4 + 255) / 256), blk, 0, stream>>>(x, xbf, (int)nx);
  cvt_f32_bf16_kernel<<<dim3((nw / 4 + 255) / 256), blk, 0, stream>>>(Win, winbf, (int)nw);
  cvt_f32_bf16_kernel<<<dim3((nw / 4 + 255) / 256), blk, 0, stream>>>(Wout, woutbf, (int)nw);
  cvt_f32_bf16_kernel<<<dim3((na / 4 + 255) / 256), blk, 0, stream>>>(aw, awbf, (int)na);

  gemm_in_kernel<<<dim3(DM / 64, (BATCH * SEQ) / 128), blk, 0, stream>>>(xbf, winbf, bin, hbuf, hTbuf);
  qproj_kernel<<<dim3(SEQ / 128, BATCH * NH), blk, 0, stream>>>(hbuf, awbf, qbuf);
  flash_attn_kernel<<<dim3(SEQ / 128, BATCH * NH), blk, 0, stream>>>(hbuf, hTbuf, qbuf, obuf);
  gemm_out_kernel<<<dim3(DM / 64, (BATCH * SEQ) / 128), blk, 0, stream>>>(obuf, woutbf, bout, out);
}
